// NeuralRoughSimulator_46969762349160
// MI455X (gfx1250) — matrix operand order repaired and checked by reading, not compiled, not run on hardware
//
#include <hip/hip_runtime.h>
#include <hip/hip_bf16.h>
#include <math.h>

// NeuralRoughSimulator for MI455X (gfx1250, wave32, WMMA).
// One wave = 16 paths x one MLP (drift or diffusion). Two waves per block
// cooperate on a path tile; weights pre-swizzled into LDS B-fragment layout.

typedef __attribute__((ext_vector_type(16))) _Float16 v16h;
typedef __attribute__((ext_vector_type(8)))  float    v8f;

#define S_STEPS 256
#define NFRAG   20      // B-fragments per net: L0:4, L1:8, L2:8
#define AIN_STR 40      // halves per row of the 15(->32)-wide input buffer
#define ACT_STR 72      // halves per row of 64-wide activation buffers (16B-aligned rows)

union V16U { v16h v; uint4 q[2]; };

__device__ __forceinline__ float fast_tanh(float x) {
#if __has_builtin(__builtin_amdgcn_tanhf)
  return __builtin_amdgcn_tanhf(x);     // gfx1250 v_tanh_f32
#else
  return tanhf(x);
#endif
}

__global__ __launch_bounds__(64)
void nrs_kernel(const float* dW0, const float* db0, const float* gW0, const float* gb0,
                const float* dW1, const float* db1, const float* gW1, const float* gb1,
                const float* dW2, const float* db2, const float* gW2, const float* gb2,
                const float* dW3, const float* db3, const float* gW3, const float* gb3,
                const float* init_var, const float* dwb, const float* dtp,
                const float* kappap, const float* thetap, float* out)
{
  __shared__ __align__(16) _Float16 s_frag[2*NFRAG*512];   // 40 KB swizzled weights (f16)
  __shared__ float s_hw[2][64];                            // head weights (f32)
  __shared__ __align__(16) _Float16 s_actIn[16*AIN_STR];   // step input features (shared)
  __shared__ __align__(16) _Float16 s_act[2][2][16*ACT_STR]; // [net][pingpong]
  __shared__ float s_raw[2][16];                           // head outputs exchange

  const float* Wh[2][3] = {{dW0,dW1,dW2},{gW0,gW1,gW2}};
  const float* Bh[2][3] = {{db0,db1,db2},{gb0,gb1,gb2}};

  // ---- one-time init: swizzle weights into WMMA B-fragment layout ----
  // B order used here (contiguous k, A matches): lanes 0-15 hold K=chunk+0..15 (elem e -> K+e), lanes 16-31 K=chunk+16..31.
  for (int idx = threadIdx.x; idx < 2*NFRAG*512; idx += 64) {
    int f = idx >> 9, e = idx & 511;
    int ln = e >> 4, h = e & 15;
    int nid = (f >= NFRAG) ? 1 : 0;
    int fl = f - nid*NFRAG;
    int l, t, c;
    if (fl < 4)       { l = 0; t = fl;           c = 0;          }
    else if (fl < 12) { l = 1; t = (fl-4) >> 1;  c = (fl-4) & 1; }
    else              { l = 2; t = (fl-12) >> 1; c = (fl-12) & 1;}
    int Kdim = l ? 64 : 15;
    int N = t*16 + (ln & 15);
    int K = c*32 + ((ln >> 4) << 4) + h;
    float w = (K < Kdim) ? Wh[nid][l][N*Kdim + K] : 0.0f;
    s_frag[idx] = (_Float16)w;
  }
  for (int k = threadIdx.x; k < 128; k += 64) {
    const float* w3 = (k >= 64) ? gW3 : dW3;
    s_hw[k >> 6][k & 63] = w3[k & 63];
  }
  for (int k = threadIdx.x; k < 16*AIN_STR; k += 64) s_actIn[k] = (_Float16)0.0f;
  __syncthreads();

  const int lane = threadIdx.x & 31;
  const int net  = threadIdx.x >> 5;   // wave 0 = drift net, wave 1 = diffusion net
  const int p    = lane & 15;          // local path
  const int hi   = lane >> 4;          // lane half
  const int path = blockIdx.x*16 + p;

  const float dt = dtp[0], kappa = kappap[0], theta = thetap[0];
  const float b3 = (net ? gb3 : db3)[0];
  float biasr[3][4];
  #pragma unroll
  for (int l = 0; l < 3; ++l)
    #pragma unroll
    for (int t = 0; t < 4; ++t)
      biasr[l][t] = Bh[net][l][t*16 + p];

  float iv = init_var[path];
  iv = fminf(1.5f, fmaxf(0.01f, iv));
  float log_v = logf(iv);
  float s1[2] = {0.f,0.f}, s2[4] = {0.f,0.f,0.f,0.f};
  float s3[8] = {0.f,0.f,0.f,0.f,0.f,0.f,0.f,0.f};

  if (net == 0 && lane < 16)
    s_actIn[p*AIN_STR + 14] = (_Float16)(log_v + 4.0f);

  // A order used here (contiguous k, matches B): lane<16 holds K {c+0..15}; lane>=16 holds {c+16..31}.
  auto loadA = [&](const _Float16* buf, int strideH, int c) -> v16h {
    int base = p*strideH + c*32 + hi*16;
    V16U u;
    u.q[0] = *(const uint4*)(buf + base);
    u.q[1] = *(const uint4*)(buf + base + 8);
    return u.v;
  };
  auto loadB = [&](int f) -> v16h {
    const _Float16* fp = s_frag + ((net*NFRAG + f) << 9) + lane*16;
    V16U u;
    u.q[0] = *(const uint4*)(fp);
    u.q[1] = *(const uint4*)(fp + 8);
    return u.v;
  };
  auto runLayer = [&](const _Float16* src, int srcStride, _Float16* dst,
                      int fbase, int nChunks, int l) {
    v16h a0 = loadA(src, srcStride, 0);
    v16h a1 = (nChunks == 2) ? loadA(src, srcStride, 1) : a0;
    #pragma unroll
    for (int t4 = 0; t4 < 4; ++t4) {
      float bv = biasr[l][t4];
      v8f c;
      #pragma unroll
      for (int r = 0; r < 8; ++r) c[r] = bv;     // bias pre-loaded into accumulator
      c = __builtin_amdgcn_wmma_f32_16x16x32_f16(false, a0, false,
            loadB(fbase + t4*nChunks), (short)0, c, false, false);
      if (nChunks == 2)
        c = __builtin_amdgcn_wmma_f32_16x16x32_f16(false, a1, false,
              loadB(fbase + t4*nChunks + 1), (short)0, c, false, false);
      // D layout: lane n -> N = n&15, VGPR r -> M = r + 8*(n>=16)
      #pragma unroll
      for (int r = 0; r < 8; ++r)
        dst[(r + hi*8)*ACT_STR + t4*16 + p] = (_Float16)fast_tanh(c[r]);
    }
  };

  _Float16* actA = s_act[net][0];
  _Float16* actB = s_act[net][1];

  for (int st = 0; st < S_STEPS; ++st) {
    __syncthreads();  // actIn from previous epilogue visible; s_raw free
    runLayer(s_actIn, AIN_STR, actA, 0, 1, 0);   // 15->64
    runLayer(actA, ACT_STR, actB, 4, 2, 1);      // 64->64
    runLayer(actB, ACT_STR, actA, 12, 2, 2);     // 64->64

    // head 64->1: each lane-half dots 32 features, combine across halves
    float sum = 0.f;
    const _Float16* row = actA + p*ACT_STR + hi*32;
    #pragma unroll
    for (int g = 0; g < 4; ++g) {
      uint4 q = *(const uint4*)(row + g*8);
      const _Float16* hh = (const _Float16*)&q;
      #pragma unroll
      for (int e = 0; e < 8; ++e)
        sum += (float)hh[e] * s_hw[net][hi*32 + g*8 + e];
    }
    sum += __shfl_xor(sum, 16);
    float raw = sum + b3;
    if (lane < 16) s_raw[net][p] = raw;
    __syncthreads();  // both nets' head outputs ready
    float raw0 = s_raw[0][p], raw1 = s_raw[1][p];

    // epilogue (redundant on both waves/halves to keep state in registers)
    float drift_nn  = 0.5f * fast_tanh(raw0);
    float diffusion = 0.75f * fast_tanh(0.5f * raw1) + 0.85f;   // 1.5*sigmoid(x)+0.1
    float dwv = dwb[path*S_STEPS + st];
    float lv = log_v + kappa*(theta - log_v)*dt + drift_nn*dt + diffusion*dwv;
    lv = fminf(1.0f, fmaxf(-5.0f, lv));
    if (net == 0 && lane < 16) out[path*S_STEPS + st] = expf(lv);

    float d = lv - log_v;
    float dx[2] = {dt, d};
    float dd[4];
    #pragma unroll
    for (int i = 0; i < 2; ++i)
      #pragma unroll
      for (int j = 0; j < 2; ++j) dd[i*2+j] = dx[i]*dx[j];
    float ns3[8], ns2[4];
    #pragma unroll
    for (int m = 0; m < 8; ++m)
      ns3[m] = s3[m] + s2[m>>1]*dx[m&1] + s1[m>>2]*(0.5f*dd[m&3])
             + dx[(m>>2)&1]*dx[(m>>1)&1]*dx[m&1]*(1.0f/6.0f);
    #pragma unroll
    for (int m = 0; m < 4; ++m)
      ns2[m] = s2[m] + s1[m>>1]*dx[m&1] + 0.5f*dd[m];
    s1[0] += dx[0]; s1[1] += dx[1];
    #pragma unroll
    for (int m = 0; m < 4; ++m) s2[m] = ns2[m];
    #pragma unroll
    for (int m = 0; m < 8; ++m) s3[m] = ns3[m];
    log_v = lv;

    // publish next step's 15 input features (cols 15..31 stay zero)
    if (net == 0 && lane < 16) {
      _Float16* r = s_actIn + p*AIN_STR;
      r[0] = (_Float16)s1[0]; r[1] = (_Float16)s1[1];
      #pragma unroll
      for (int m = 0; m < 4; ++m) r[2+m] = (_Float16)s2[m];
      #pragma unroll
      for (int m = 0; m < 8; ++m) r[6+m] = (_Float16)s3[m];
      r[14] = (_Float16)(log_v + 4.0f);
    }
  }
}

extern "C" void kernel_launch(void* const* d_in, const int* in_sizes, int n_in,
                              void* d_out, int out_size, void* d_ws, size_t ws_size,
                              hipStream_t stream) {
  (void)n_in; (void)d_ws; (void)ws_size; (void)out_size;
  const float* P[21];
  for (int i = 0; i < 21; ++i) P[i] = (const float*)d_in[i];
  const int B = in_sizes[16];          // init_var length = 2048 paths
  dim3 grid((B + 15) / 16), block(64); // 16 paths per block, 2 waves (drift+diff)
  nrs_kernel<<<grid, block, 0, stream>>>(
      P[0], P[1], P[2], P[3],     // dW0, db0, gW0, gb0
      P[4], P[5], P[6], P[7],     // dW1, db1, gW1, gb1
      P[8], P[9], P[10], P[11],   // dW2, db2, gW2, gb2
      P[12], P[13], P[14], P[15], // dW3, db3, gW3, gb3
      P[16], P[17], P[18], P[19], P[20],  // init_var, dw, dt, kappa, theta
      (float*)d_out);
}
